// IMAEN_36850819400243
// MI455X (gfx1250) — matrix operand order repaired and checked by reading, not compiled, not run on hardware
//
#include <hip/hip_runtime.h>
#include <hip/hip_bf16.h>

typedef _Float16 h16;
typedef __attribute__((ext_vector_type(16))) _Float16 v16h;
typedef __attribute__((ext_vector_type(8)))  float    v8f;
typedef unsigned int u32x4 __attribute__((ext_vector_type(4)));
typedef int          i32x8 __attribute__((ext_vector_type(8)));
typedef int          i32x4 __attribute__((ext_vector_type(4)));

#define BATCH 512
#define NNODE 32
#define BN_ROWS (BATCH * NNODE)   // 16384
#define LSEQ 1000
#define CHUNK 64                  // protein batch chunk

// Tensor Data Mover path (clang-23 6-arg builtin), with fallback.
#if defined(__HIP_DEVICE_COMPILE__)
#if __has_builtin(__builtin_amdgcn_tensor_load_to_lds) && __has_builtin(__builtin_amdgcn_s_wait_tensorcnt)
#define ATHENA_TDM 1
#endif
#endif

// ---------------------------------------------------------------------------
// WMMA GEMM: C[M,N] = act(A[M,K] @ Bt^T + bias).
// A: f32 [M,K], K % 32 == 0, M % 32 == 0 (padded by producers).
// Bt: f16 [N,K] (pre-transposed, K zero-padded) read directly from global.
// block = 128 threads = 4 waves; tile = 32(M) x 64(N); 2 accumulators per wave
// so each B fragment feeds 2 WMMAs. A double-buffered in LDS.
// ---------------------------------------------------------------------------
__global__ __launch_bounds__(128)
void gemm_wmma(const float* __restrict__ A, const h16* __restrict__ Bt,
               const float* __restrict__ bias, float* __restrict__ C,
               int M, int N, int K, int act)
{
    __shared__ h16 As[2][32 * 32];
    const int tid  = threadIdx.x;
    const int wave = tid >> 5;
    const int lane = tid & 31;
    const int m0 = blockIdx.x * 32;
    const int n0 = blockIdx.y * 64;
    const int mrow = lane & 15;
    const int lhi  = lane >> 4;
    const int ng  = n0 + wave * 16 + mrow;
    const int ngc = (ng < N) ? ng : (N - 1);
    const h16* browK = Bt + (size_t)ngc * K + (lhi << 4);

    // A staging: 32 rows x 8 quads of 4 floats; each thread covers rows r, r+16
    const int r  = tid >> 3;
    const int cq = (tid & 7) << 2;
    const float* aptr0 = A + (size_t)(m0 + r) * K + cq;
    const float* aptr1 = aptr0 + (size_t)16 * K;

    {   // stage k0 = 0 into buffer 0
        float4 v0 = *(const float4*)aptr0;
        float4 v1 = *(const float4*)aptr1;
        h16* d0 = &As[0][r * 32 + cq];
        h16* d1 = &As[0][(r + 16) * 32 + cq];
        d0[0] = (h16)v0.x; d0[1] = (h16)v0.y; d0[2] = (h16)v0.z; d0[3] = (h16)v0.w;
        d1[0] = (h16)v1.x; d1[1] = (h16)v1.y; d1[2] = (h16)v1.z; d1[3] = (h16)v1.w;
    }
    v8f acc0 = {}, acc1 = {};
    int cur = 0;
    for (int k0 = 0; k0 < K; k0 += 32, cur ^= 1) {
        __syncthreads();
        if (k0 + 32 < K) {      // prefetch next K-tile into other buffer
            float4 v0 = *(const float4*)(aptr0 + k0 + 32);
            float4 v1 = *(const float4*)(aptr1 + k0 + 32);
            h16* d0 = &As[cur ^ 1][r * 32 + cq];
            h16* d1 = &As[cur ^ 1][(r + 16) * 32 + cq];
            d0[0] = (h16)v0.x; d0[1] = (h16)v0.y; d0[2] = (h16)v0.z; d0[3] = (h16)v0.w;
            d1[0] = (h16)v1.x; d1[1] = (h16)v1.y; d1[2] = (h16)v1.z; d1[3] = (h16)v1.w;
        }
        v16h a0, a1, b;
        const h16* arow0 = &As[cur][mrow * 32];
        const h16* arow1 = &As[cur][(16 + mrow) * 32];
        #pragma unroll
        for (int j = 0; j < 8; ++j) {
            // A layout (matches B): VGPR j holds K pair (j*2 + lhi*16)
            int kb = (j << 1) + (lhi << 4);
            a0[2 * j]     = arow0[kb];
            a0[2 * j + 1] = arow0[kb + 1];
            a1[2 * j]     = arow1[kb];
            a1[2 * j + 1] = arow1[kb + 1];
        }
        const h16* bp = browK + k0;          // 32B contiguous per lane
        #pragma unroll
        for (int e = 0; e < 16; ++e) b[e] = bp[e];
        acc0 = __builtin_amdgcn_wmma_f32_16x16x32_f16(false, a0, false, b,
                                                      (short)0, acc0, false, false);
        acc1 = __builtin_amdgcn_wmma_f32_16x16x32_f16(false, a1, false, b,
                                                      (short)0, acc1, false, false);
    }
    if (ng < N) {
        float bv = bias ? bias[ng] : 0.f;
        #pragma unroll
        for (int rr = 0; rr < 8; ++rr) {
            int mg0 = m0 + rr + (lhi << 3);
            float v0 = acc0[rr] + bv;
            float v1 = acc1[rr] + bv;
            if (act == 1) { v0 = fmaxf(v0, 0.f); v1 = fmaxf(v1, 0.f); }
            C[(size_t)mg0 * N + ng]        = v0;
            C[(size_t)(mg0 + 16) * N + ng] = v1;
        }
    }
}

// ---------------------------------------------------------------------------
// Implicit-GEMM K=3 valid conv + relu.
// in : [CB][Tin][Cin] f16 (Cin in {96,128}); Wt: f16 [96][3][Cin]; out relu'd f16.
// block = 96 threads = 3 waves; tile = 32(t) x 48(o); 2 accumulators per wave.
// Activation window (34 rows x Cin) staged once per block — via TDM if available.
// ---------------------------------------------------------------------------
__global__ __launch_bounds__(96)
void conv3_wmma(const h16* __restrict__ in, const h16* __restrict__ Wt,
                const float* __restrict__ bias, h16* __restrict__ out,
                int Tin, int Cin)
{
    const int Tout = Tin - 2;
    __shared__ h16 As[34 * 128];
    const int tid  = threadIdx.x;
    const int wave = tid >> 5;
    const int lane = tid & 31;
    const int t0 = blockIdx.x * 32;
    const int ob = blockIdx.y * 48;
    const int b  = blockIdx.z;
    const int mrow = lane & 15;
    const int lhi  = lane >> 4;
    const h16* src = in + (size_t)b * Tin * Cin + (size_t)t0 * Cin;

#if defined(ATHENA_TDM)
    if (wave == 0) {
        unsigned long long ga = (unsigned long long)(uintptr_t)src;
        unsigned int lds = (unsigned int)(uintptr_t)(&As[0]);
        unsigned int rows = (unsigned int)(Tin - t0);     // OOB rows read as zero
        u32x4 g0; i32x8 g1;
        i32x4 g2 = {0, 0, 0, 0}; i32x4 g3 = {0, 0, 0, 0};
        i32x8 g4 = {0, 0, 0, 0, 0, 0, 0, 0};
        g0[0] = 1u;                                       // count=1 valid descriptor
        g0[1] = lds;                                      // lds_addr
        g0[2] = (unsigned int)ga;                         // global_addr lo
        g0[3] = ((unsigned int)(ga >> 32) & 0x01FFFFFFu) | (2u << 30);  // hi | type=2
        g1[0] = (int)(1u << 16);                          // data_size = 2 bytes
        g1[1] = (int)(((unsigned)Cin & 0xFFFFu) << 16);   // tensor_dim0 lo16
        g1[2] = (int)((rows & 0xFFFFu) << 16);            // dim0 hi | tensor_dim1 lo16
        g1[3] = (int)(((rows >> 16) & 0xFFFFu) | ((unsigned)Cin << 16)); // dim1 hi | tile_dim0
        g1[4] = 34;                                       // tile_dim1=34, tile_dim2=0
        g1[5] = Cin;                                      // tensor_dim0_stride lo32
        g1[6] = 0;
        g1[7] = 0;
        __builtin_amdgcn_tensor_load_to_lds(g0, g1, g2, g3, g4, 0);
        __builtin_amdgcn_s_wait_tensorcnt(0);
    }
    __syncthreads();
#else
    {
        const unsigned int* s32 = (const unsigned int*)src;
        unsigned int* d32 = (unsigned int*)As;
        const int nw = Cin >> 1;                          // dwords per row
        for (int i = tid; i < 34 * nw; i += 96) {
            int rr = i / nw;
            d32[i] = (t0 + rr < Tin) ? s32[i] : 0u;
        }
    }
    __syncthreads();
#endif

    const h16* brow = Wt + (size_t)(ob + wave * 16 + mrow) * (3 * Cin) + (lhi << 4);
    v8f acc0 = {}, acc1 = {};
    for (int c0 = 0; c0 < Cin; c0 += 32) {
        #pragma unroll
        for (int k = 0; k < 3; ++k) {
            v16h a0, a1, bb;
            const h16* arow0 = &As[(mrow + k) * Cin + c0];
            const h16* arow1 = &As[(16 + mrow + k) * Cin + c0];
            #pragma unroll
            for (int j = 0; j < 8; ++j) {
                int kb = (j << 1) + (lhi << 4);
                a0[2 * j]     = arow0[kb];
                a0[2 * j + 1] = arow0[kb + 1];
                a1[2 * j]     = arow1[kb];
                a1[2 * j + 1] = arow1[kb + 1];
            }
            const h16* bp = brow + k * Cin + c0;
            #pragma unroll
            for (int e = 0; e < 16; ++e) bb[e] = bp[e];
            acc0 = __builtin_amdgcn_wmma_f32_16x16x32_f16(false, a0, false, bb,
                                                          (short)0, acc0, false, false);
            acc1 = __builtin_amdgcn_wmma_f32_16x16x32_f16(false, a1, false, bb,
                                                          (short)0, acc1, false, false);
        }
    }
    int og = ob + wave * 16 + mrow;
    float bv = bias[og];
    #pragma unroll
    for (int rr = 0; rr < 8; ++rr) {
        int t  = t0 + rr + (lhi << 3);
        int t2 = t + 16;
        if (t < Tout) {
            float v = acc0[rr] + bv;
            out[((size_t)b * Tout + t) * 96 + og] = (h16)fmaxf(v, 0.f);
        }
        if (t2 < Tout) {
            float v = acc1[rr] + bv;
            out[((size_t)b * Tout + t2) * 96 + og] = (h16)fmaxf(v, 0.f);
        }
    }
}

// ---------------------------------------------------------------------------
// Weight preconversion
// ---------------------------------------------------------------------------
// Bt[n][k] = (k < K) ? B[k][n] : 0   (f32 -> f16, K padded to Kp)
__global__ void conv_bt(const float* __restrict__ B, h16* __restrict__ Bt,
                        int K, int N, int Kp)
{
    size_t i = (size_t)blockIdx.x * 256 + threadIdx.x;
    if (i >= (size_t)N * Kp) return;
    int n = (int)(i / Kp), k = (int)(i % Kp);
    Bt[i] = (k < K) ? (h16)B[(size_t)k * N + n] : (h16)0.f;
}

// Wt[o][k][i] = W[o][i][k]  (f32 -> f16), conv weights [96][Cin][3]
__global__ void conv_wt(const float* __restrict__ W, h16* __restrict__ Wt, int Cin)
{
    int total = 96 * 3 * Cin;
    int i = blockIdx.x * 256 + threadIdx.x;
    if (i >= total) return;
    int o = i / (3 * Cin); int rem = i - o * 3 * Cin;
    int k = rem / Cin;     int ii = rem - k * Cin;
    Wt[i] = (h16)W[(size_t)o * Cin * 3 + (size_t)ii * 3 + k];
}

// ---------------------------------------------------------------------------
// Elementwise / reduction helpers
// ---------------------------------------------------------------------------
__global__ void pack2(const float* __restrict__ A, int Ca,
                      const float* __restrict__ B, int Cb,
                      float* __restrict__ O, int Mrows, int Cout)  // zero-pads c >= Ca+Cb
{
    size_t i = (size_t)blockIdx.x * 256 + threadIdx.x;
    if (i >= (size_t)Mrows * Cout) return;
    int r = (int)(i / Cout), c = (int)(i - (size_t)r * Cout);
    float v = 0.f;
    if (c < Ca)            v = A[(size_t)r * Ca + c];
    else if (c < Ca + Cb)  v = B[(size_t)r * Cb + (c - Ca)];
    O[i] = v;
}

__global__ void bn_stats(const float* __restrict__ X, int Mrows, int C,
                         float* __restrict__ mean, float* __restrict__ var)
{
    __shared__ float s1[256], s2[256];
    int c = blockIdx.x;
    float a = 0.f, b = 0.f;
    for (int r = threadIdx.x; r < Mrows; r += 256) {
        float v = X[(size_t)r * C + c];
        a += v; b += v * v;
    }
    s1[threadIdx.x] = a; s2[threadIdx.x] = b;
    __syncthreads();
    for (int s = 128; s > 0; s >>= 1) {
        if (threadIdx.x < s) { s1[threadIdx.x] += s1[threadIdx.x + s];
                               s2[threadIdx.x] += s2[threadIdx.x + s]; }
        __syncthreads();
    }
    if (threadIdx.x == 0) {
        float m = s1[0] / (float)Mrows;
        mean[c] = m;
        var[c]  = s2[0] / (float)Mrows - m * m;
    }
}

__global__ void bn_apply(const float* __restrict__ X, const float* __restrict__ mean,
                         const float* __restrict__ var, const float* __restrict__ g,
                         const float* __restrict__ be, float* __restrict__ Y,
                         int Mrows, int C, int act)   // 1 relu, 2 tanh
{
    size_t i = (size_t)blockIdx.x * 256 + threadIdx.x;
    if (i >= (size_t)Mrows * C) return;
    int c = (int)(i % C);
    float v = (X[i] - mean[c]) * rsqrtf(var[c] + 1e-5f) * g[c] + be[c];
    if (act == 1) v = fmaxf(v, 0.f);
    else if (act == 2) v = tanhf(v);
    Y[i] = v;
}

__global__ void adj_norm3(const float* __restrict__ adj, float* __restrict__ A1,
                          float* __restrict__ A2, float* __restrict__ A3)
{
    __shared__ float sAdj[1024], sP[1024], sQ[1024];
    __shared__ float sDinv[32];
    const int b = blockIdx.x, tid = threadIdx.x;
    const float* ab = adj + (size_t)b * 1024;
    for (int i = tid; i < 1024; i += 256) sAdj[i] = ab[i];
    __syncthreads();
    if (tid < 32) {
        float s = 0.f;
        for (int j = 0; j < 32; ++j) { float v = sAdj[tid * 32 + j]; if (j == tid) v = fmaxf(v, 1.f); s += v; }
        sDinv[tid] = rsqrtf(s);
    }
    __syncthreads();
    for (int i = tid; i < 1024; i += 256) {
        int r = i >> 5, c = i & 31;
        float v = sAdj[i]; if (r == c) v = fmaxf(v, 1.f);
        A1[(size_t)b * 1024 + i] = v * sDinv[r] * sDinv[c];
    }
    for (int i = tid; i < 1024; i += 256) {
        int r = i >> 5, c = i & 31;
        float s = 0.f;
        for (int k = 0; k < 32; ++k) s += sAdj[r * 32 + k] * sAdj[k * 32 + c];
        sP[i] = (s > 0.f) ? 1.f : 0.f;
    }
    __syncthreads();
    if (tid < 32) {
        float s = 0.f;
        for (int j = 0; j < 32; ++j) { float v = sP[tid * 32 + j]; if (j == tid) v = fmaxf(v, 1.f); s += v; }
        sDinv[tid] = rsqrtf(s);
    }
    __syncthreads();
    for (int i = tid; i < 1024; i += 256) {
        int r = i >> 5, c = i & 31;
        float v = sP[i]; if (r == c) v = fmaxf(v, 1.f);
        A2[(size_t)b * 1024 + i] = v * sDinv[r] * sDinv[c];
    }
    for (int i = tid; i < 1024; i += 256) {
        int r = i >> 5, c = i & 31;
        float s = 0.f;
        for (int k = 0; k < 32; ++k) s += sP[r * 32 + k] * sAdj[k * 32 + c];
        sQ[i] = (s > 0.f) ? 1.f : 0.f;
    }
    __syncthreads();
    if (tid < 32) {
        float s = 0.f;
        for (int j = 0; j < 32; ++j) { float v = sQ[tid * 32 + j]; if (j == tid) v = fmaxf(v, 1.f); s += v; }
        sDinv[tid] = rsqrtf(s);
    }
    __syncthreads();
    for (int i = tid; i < 1024; i += 256) {
        int r = i >> 5, c = i & 31;
        float v = sQ[i]; if (r == c) v = fmaxf(v, 1.f);
        A3[(size_t)b * 1024 + i] = v * sDinv[r] * sDinv[c];
    }
}

// Y[b][:, c<C] = relu(A[b] @ H[b] + bias); zero-fills C <= c < Cp (Y stride = Cp).
__global__ void gcn_aggregate(const float* __restrict__ Ab, const float* __restrict__ H,
                              const float* __restrict__ bias, float* __restrict__ Y,
                              int C, int Cp)
{
    __shared__ float sA[1024];
    const int b = blockIdx.x;
    const int c0 = blockIdx.y * 32;
    const int tid = threadIdx.x;
    for (int i = tid; i < 1024; i += 256) sA[i] = Ab[(size_t)b * 1024 + i];
    __syncthreads();
    int c = c0 + (tid & 31);
    int rb = tid >> 5;
    if (c >= Cp) return;
    float* Yb = Y + (size_t)b * 32 * Cp;
    if (c >= C) {
        Yb[(size_t)rb * Cp + c] = 0.f;        Yb[(size_t)(rb + 8) * Cp + c] = 0.f;
        Yb[(size_t)(rb + 16) * Cp + c] = 0.f; Yb[(size_t)(rb + 24) * Cp + c] = 0.f;
        return;
    }
    float a0 = 0.f, a1 = 0.f, a2 = 0.f, a3 = 0.f;
    const float* Hb = H + (size_t)b * 32 * C;
    for (int k = 0; k < 32; ++k) {
        float h = Hb[(size_t)k * C + c];
        a0 += sA[rb * 32 + k] * h;
        a1 += sA[(rb + 8) * 32 + k] * h;
        a2 += sA[(rb + 16) * 32 + k] * h;
        a3 += sA[(rb + 24) * 32 + k] * h;
    }
    float bv = bias[c];
    Yb[(size_t)rb * Cp + c]        = fmaxf(a0 + bv, 0.f);
    Yb[(size_t)(rb + 8) * Cp + c]  = fmaxf(a1 + bv, 0.f);
    Yb[(size_t)(rb + 16) * Cp + c] = fmaxf(a2 + bv, 0.f);
    Yb[(size_t)(rb + 24) * Cp + c] = fmaxf(a3 + bv, 0.f);
}

// hg[b,f] = max_n concat(h3|h5|h6)[b,n,f]; hg stride 1120, zero-padded tail
__global__ void gmp_concat(const float* __restrict__ h3, const float* __restrict__ h5,
                           const float* __restrict__ h6, float* __restrict__ hg)
{
    int b = blockIdx.x;
    int f = blockIdx.y * 256 + threadIdx.x;
    if (f >= 1120) return;
    if (f >= 1092) { hg[(size_t)b * 1120 + f] = 0.f; return; }
    const float* src; int C, off;
    if (f < 624)      { src = h3; C = 624; off = f; }
    else if (f < 936) { src = h5; C = 312; off = f - 624; }
    else              { src = h6; C = 156; off = f - 936; }
    const float* p = src + (size_t)b * 32 * C + off;
    float m = -1e30f;
    for (int n = 0; n < 32; ++n) m = fmaxf(m, p[(size_t)n * C]);
    hg[(size_t)b * 1120 + f] = m;
}

__global__ void attn_softmax(const float* __restrict__ V, float* __restrict__ dst,
                             int dstStride, int dstOff)
{
    __shared__ float red[128];
    int r = blockIdx.x, t = threadIdx.x;
    float v = V[(size_t)r * 128 + t];
    float tv = tanhf(v);
    red[t] = tv; __syncthreads();
    for (int s = 64; s > 0; s >>= 1) { if (t < s) red[t] = fmaxf(red[t], red[t + s]); __syncthreads(); }
    float mx = red[0]; __syncthreads();
    float e = expf(tv - mx);
    red[t] = e; __syncthreads();
    for (int s = 64; s > 0; s >>= 1) { if (t < s) red[t] += red[t + s]; __syncthreads(); }
    float sum = red[0];
    dst[(size_t)r * dstStride + dstOff + t] = (e / sum) * v;
}

__global__ void emb_gather(const int* __restrict__ target, const float* __restrict__ emb,
                           h16* __restrict__ E, int bstart)
{
    size_t i = (size_t)blockIdx.x * 256 + threadIdx.x;
    if (i >= (size_t)CHUNK * LSEQ * 128) return;
    int ch = (int)(i & 127);
    size_t rem = i >> 7;
    int t  = (int)(rem % LSEQ);
    int bl = (int)(rem / LSEQ);
    int row = target[(size_t)(bstart + bl) * LSEQ + t];
    E[i] = (h16)emb[row * 128 + ch];
}

__global__ void max_over_t(const h16* __restrict__ buf, int Tout,
                           float* __restrict__ f123, int bstart, int sec)
{
    int bl = blockIdx.x, o = threadIdx.x;   // 96 threads
    const h16* p = buf + (size_t)bl * Tout * 96 + o;
    float m = -1e30f;
    for (int t = 0; t < Tout; ++t) m = fmaxf(m, (float)p[(size_t)t * 96]);
    f123[(size_t)(bstart + bl) * 288 + sec * 96 + o] = m;
}

// ---------------------------------------------------------------------------
extern "C" void kernel_launch(void* const* d_in, const int* in_sizes, int n_in,
                              void* d_out, int out_size, void* d_ws, size_t ws_size,
                              hipStream_t stream)
{
    (void)in_sizes; (void)n_in; (void)out_size; (void)ws_size;
    const float* x       = (const float*)d_in[0];
    const float* adj     = (const float*)d_in[1];
    const int*   target  = (const int*)  d_in[2];
    const float* z       = (const float*)d_in[3];
    const float* dec_W0  = (const float*)d_in[4];
    const float* dec_b0  = (const float*)d_in[5];
    const float* dec_g0  = (const float*)d_in[6];
    const float* dec_be0 = (const float*)d_in[7];
    const float* dec_W1  = (const float*)d_in[8];
    const float* dec_b1  = (const float*)d_in[9];
    const float* dec_g1  = (const float*)d_in[10];
    const float* dec_be1 = (const float*)d_in[11];
    const float* gW1 = (const float*)d_in[12]; const float* gb1 = (const float*)d_in[13];
    const float* gW2 = (const float*)d_in[14]; const float* gb2 = (const float*)d_in[15];
    const float* gW3 = (const float*)d_in[16]; const float* gb3 = (const float*)d_in[17];
    const float* fcg1_W = (const float*)d_in[18]; const float* fcg1_b = (const float*)d_in[19];
    const float* fcg2_W = (const float*)d_in[20]; const float* fcg2_b = (const float*)d_in[21];
    const float* emb    = (const float*)d_in[22];
    const float* c10_W = (const float*)d_in[23]; const float* c10_b = (const float*)d_in[24];
    const float* c20_W = (const float*)d_in[25]; const float* c20_b = (const float*)d_in[26];
    const float* c21_W = (const float*)d_in[27]; const float* c21_b = (const float*)d_in[28];
    const float* c30_W = (const float*)d_in[29]; const float* c30_b = (const float*)d_in[30];
    const float* c31_W = (const float*)d_in[31]; const float* c31_b = (const float*)d_in[32];
    const float* c32_W = (const float*)d_in[33]; const float* c32_b = (const float*)d_in[34];
    const float* prot_W = (const float*)d_in[35]; const float* prot_b = (const float*)d_in[36];
    const float* fc1_W = (const float*)d_in[37]; const float* fc1_b = (const float*)d_in[38];
    const float* fc2_W = (const float*)d_in[39]; const float* fc2_b = (const float*)d_in[40];
    const float* out_W = (const float*)d_in[41]; const float* out_b = (const float*)d_in[42];

    // -------- workspace carve --------
    char* wptr = (char*)d_ws;
    auto alloc_f = [&](size_t n) -> float* {
        float* p = (float*)wptr; wptr += ((n * 4) + 255) & ~(size_t)255; return p;
    };
    auto alloc_h = [&](size_t n) -> h16* {
        h16* p = (h16*)wptr; wptr += ((n * 2) + 255) & ~(size_t)255; return p;
    };
    const int Bn = BN_ROWS;
    float* zc   = alloc_f((size_t)Bn * 160);
    float* h0   = alloc_f((size_t)Bn * 256);
    float* p1   = alloc_f((size_t)Bn * 78);
    float* mb0  = alloc_f(256); float* vb0 = alloc_f(256);
    float* mb1  = alloc_f(78);  float* vb1 = alloc_f(78);
    float* X    = alloc_f((size_t)Bn * 160);
    float* A1   = alloc_f((size_t)BATCH * 1024);
    float* A2   = alloc_f((size_t)BATCH * 1024);
    float* A3   = alloc_f((size_t)BATCH * 1024);
    float* xw1  = alloc_f((size_t)Bn * 156);
    float* h1   = alloc_f((size_t)Bn * 160);
    float* t5   = alloc_f((size_t)Bn * 160);
    float* h6   = alloc_f((size_t)Bn * 156);
    float* t312 = alloc_f((size_t)Bn * 312);
    float* h2   = alloc_f((size_t)Bn * 320);
    float* h5   = alloc_f((size_t)Bn * 312);
    float* t624 = alloc_f((size_t)Bn * 624);
    float* h3   = alloc_f((size_t)Bn * 624);
    float* hg   = alloc_f((size_t)BATCH * 1120);
    float* fg1  = alloc_f((size_t)BATCH * 1024);
    float* xgpre= alloc_f((size_t)BATCH * 128);
    float* f123 = alloc_f((size_t)BATCH * 288);
    float* xtpre= alloc_f((size_t)BATCH * 128);
    float* xc   = alloc_f((size_t)BATCH * 256);
    float* fcA  = alloc_f((size_t)BATCH * 1024);
    float* fcB  = alloc_f((size_t)BATCH * 512);
    // pre-transposed f16 weights (Bt[n][k], K padded to /32)
    h16* BtW0  = alloc_h((size_t)256 * 160);
    h16* BtW1  = alloc_h((size_t)78 * 256);
    h16* BtG1  = alloc_h((size_t)156 * 160);
    h16* BtG2  = alloc_h((size_t)312 * 160);
    h16* BtG3  = alloc_h((size_t)624 * 320);
    h16* BtFg1 = alloc_h((size_t)1024 * 1120);
    h16* BtFg2 = alloc_h((size_t)128 * 1024);
    h16* BtPr  = alloc_h((size_t)128 * 288);
    h16* BtFc1 = alloc_h((size_t)1024 * 256);
    h16* BtFc2 = alloc_h((size_t)512 * 1024);
    h16* BtOut = alloc_h((size_t)1 * 512);
    h16* Wt10 = alloc_h(96 * 384); h16* Wt20 = alloc_h(96 * 384); h16* Wt30 = alloc_h(96 * 384);
    h16* Wt21 = alloc_h(96 * 288); h16* Wt31 = alloc_h(96 * 288); h16* Wt32 = alloc_h(96 * 288);
    h16* Eh = alloc_h((size_t)CHUNK * LSEQ * 128);
    h16* cA = alloc_h((size_t)CHUNK * 998 * 96);
    h16* cB = alloc_h((size_t)CHUNK * 996 * 96);

    auto bt = [&](const float* B, h16* Bd, int K, int N, int Kp) {
        size_t tot = (size_t)N * Kp;
        conv_bt<<<(tot + 255) / 256, 256, 0, stream>>>(B, Bd, K, N, Kp);
    };
    auto wtc = [&](const float* W, h16* Wd, int Cin) {
        int tot = 96 * 3 * Cin;
        conv_wt<<<(tot + 255) / 256, 256, 0, stream>>>(W, Wd, Cin);
    };
    auto gemm = [&](const float* Aa, const h16* Bb, const float* bias, float* Cc,
                    int M, int N, int K, int act) {
        dim3 g((M + 31) / 32, (N + 63) / 64, 1);
        gemm_wmma<<<g, 128, 0, stream>>>(Aa, Bb, bias, Cc, M, N, K, act);
    };
    auto agg = [&](const float* Ab, const float* H, const float* bias, float* Y,
                   int C, int Cp) {
        dim3 g(BATCH, (Cp + 31) / 32, 1);
        gcn_aggregate<<<g, 256, 0, stream>>>(Ab, H, bias, Y, C, Cp);
    };
    auto conv = [&](const h16* in, const h16* Wd, const float* bias, h16* out,
                    int Tin, int Cin) {
        int Tout = Tin - 2;
        dim3 g((Tout + 31) / 32, 2, CHUNK);
        conv3_wmma<<<g, 96, 0, stream>>>(in, Wd, bias, out, Tin, Cin);
    };

    // ---- weight preconversion (f32 -> f16, transpose, K-pad) ----
    bt(dec_W0, BtW0, 142, 256, 160);   bt(dec_W1, BtW1, 256, 78, 256);
    bt(gW1, BtG1, 156, 156, 160);      bt(gW2, BtG2, 156, 312, 160);
    bt(gW3, BtG3, 312, 624, 320);      bt(fcg1_W, BtFg1, 1092, 1024, 1120);
    bt(fcg2_W, BtFg2, 1024, 128, 1024);bt(prot_W, BtPr, 288, 128, 288);
    bt(fc1_W, BtFc1, 256, 1024, 256);  bt(fc2_W, BtFc2, 1024, 512, 1024);
    bt(out_W, BtOut, 512, 1, 512);
    wtc(c10_W, Wt10, 128); wtc(c20_W, Wt20, 128); wtc(c30_W, Wt30, 128);
    wtc(c21_W, Wt21, 96);  wtc(c31_W, Wt31, 96);  wtc(c32_W, Wt32, 96);

    // ---- CVAE decoder ----
    pack2<<<((size_t)Bn * 160 + 255) / 256, 256, 0, stream>>>(z, 64, x, 78, zc, Bn, 160);
    gemm(zc, BtW0, dec_b0, h0, Bn, 256, 160, 0);
    bn_stats<<<256, 256, 0, stream>>>(h0, Bn, 256, mb0, vb0);
    bn_apply<<<((size_t)Bn * 256 + 255) / 256, 256, 0, stream>>>(h0, mb0, vb0, dec_g0, dec_be0, h0, Bn, 256, 1);
    gemm(h0, BtW1, dec_b1, p1, Bn, 78, 256, 0);
    bn_stats<<<78, 256, 0, stream>>>(p1, Bn, 78, mb1, vb1);
    bn_apply<<<((size_t)Bn * 78 + 255) / 256, 256, 0, stream>>>(p1, mb1, vb1, dec_g1, dec_be1, p1, Bn, 78, 2);
    pack2<<<((size_t)Bn * 160 + 255) / 256, 256, 0, stream>>>(p1, 78, x, 78, X, Bn, 160);

    // ---- adjacency norms ----
    adj_norm3<<<BATCH, 256, 0, stream>>>(adj, A1, A2, A3);

    // ---- GCN tower ----
    gemm(X, BtG1, nullptr, xw1, Bn, 156, 160, 0);
    agg(A1, xw1, gb1, h1, 156, 160);
    agg(A2, xw1, gb1, t5, 156, 160);
    agg(A3, xw1, gb1, h6, 156, 156);
    gemm(h1, BtG2, nullptr, t312, Bn, 312, 160, 0);  agg(A1, t312, gb2, h2, 312, 320);
    gemm(t5, BtG2, nullptr, t312, Bn, 312, 160, 0);  agg(A2, t312, gb2, h5, 312, 312);
    gemm(h2, BtG3, nullptr, t624, Bn, 624, 320, 0);  agg(A1, t624, gb3, h3, 624, 624);
    gmp_concat<<<dim3(BATCH, 5, 1), 256, 0, stream>>>(h3, h5, h6, hg);
    gemm(hg, BtFg1, fcg1_b, fg1, BATCH, 1024, 1120, 1);
    gemm(fg1, BtFg2, fcg2_b, xgpre, BATCH, 128, 1024, 0);
    attn_softmax<<<BATCH, 128, 0, stream>>>(xgpre, xc, 256, 0);

    // ---- protein conv branch, chunked ----
    for (int ch = 0; ch < BATCH / CHUNK; ++ch) {
        int bs = ch * CHUNK;
        emb_gather<<<((size_t)CHUNK * LSEQ * 128 + 255) / 256, 256, 0, stream>>>(target, emb, Eh, bs);
        conv(Eh, Wt10, c10_b, cA, 1000, 128);
        max_over_t<<<CHUNK, 96, 0, stream>>>(cA, 998, f123, bs, 0);
        conv(Eh, Wt20, c20_b, cA, 1000, 128);
        conv(cA, Wt21, c21_b, cB, 998, 96);
        max_over_t<<<CHUNK, 96, 0, stream>>>(cB, 996, f123, bs, 1);
        conv(Eh, Wt30, c30_b, cA, 1000, 128);
        conv(cA, Wt31, c31_b, cB, 998, 96);
        conv(cB, Wt32, c32_b, cA, 996, 96);
        max_over_t<<<CHUNK, 96, 0, stream>>>(cA, 994, f123, bs, 2);
    }
    gemm(f123, BtPr, prot_b, xtpre, BATCH, 128, 288, 0);
    attn_softmax<<<BATCH, 128, 0, stream>>>(xtpre, xc, 256, 128);

    // ---- head MLP ----
    gemm(xc, BtFc1, fc1_b, fcA, BATCH, 1024, 256, 1);
    gemm(fcA, BtFc2, fc2_b, fcB, BATCH, 512, 1024, 1);
    gemm(fcB, BtOut, out_b, (float*)d_out, BATCH, 1, 512, 0);
}
